// sLSTMBlock_72911364817333
// MI455X (gfx1250) — compile-verified, matrix operand order checked by reading, not run on hardware
//
#include <hip/hip_runtime.h>
#include <math.h>
#include <stdint.h>

// ---------------------------------------------------------------------------
// sLSTM block for MI455X (gfx1250): bf16 WMMA GEMMs (M=32 wave tiles),
// on-chip recurrent scan with async-to-LDS double-buffered gx prefetch.
// ---------------------------------------------------------------------------

#define Bdim 8
#define Tdim 2048
#define Ddim 1024
#define NH   4
#define DH   256
#define Fdim 2752
#define NT   (Bdim * Tdim)   // 16384 tokens

typedef __attribute__((ext_vector_type(16))) __bf16 v16bf;
typedef __attribute__((ext_vector_type(8)))  float  v8f;

__device__ __forceinline__ unsigned short f2bf(float f) {
  union { float f; unsigned u; } v; v.f = f;
  unsigned r = v.u + 0x7FFFu + ((v.u >> 16) & 1u);   // round-to-nearest-even
  return (unsigned short)(r >> 16);
}

__device__ __forceinline__ v8f zero8() {
  v8f z;
#pragma unroll
  for (int i = 0; i < 8; ++i) z[i] = 0.0f;
  return z;
}

// Fragment loader: A (rows = M) or B stored "N-major, K-contiguous"
// (mem[n*ld + k]).  One 32B contiguous load per lane.
__device__ __forceinline__ v16bf frag16(const unsigned short* base, int row0,
                                        int ld, int k0) {
  const int lane = threadIdx.x & 31;
  const unsigned short* p =
      base + (size_t)(row0 + (lane & 15)) * ld + k0 + ((lane >> 4) << 4);
  return *reinterpret_cast<const v16bf*>(p);
}

// ---------------------------------------------------------------------------
// fp32 -> bf16 converters (one-time weight prep)
// ---------------------------------------------------------------------------
__global__ void k_cvt_bf16(const float* __restrict__ in,
                           unsigned short* __restrict__ out, int n) {
  for (int i = blockIdx.x * blockDim.x + threadIdx.x; i < n;
       i += gridDim.x * blockDim.x)
    out[i] = f2bf(in[i]);
}

// out[c*rows + r] = bf16(in[r*cols + c])   (transpose while converting)
__global__ void k_cvt_bf16_T(const float* __restrict__ in,
                             unsigned short* __restrict__ out, int rows,
                             int cols) {
  int n = rows * cols;
  for (int i = blockIdx.x * blockDim.x + threadIdx.x; i < n;
       i += gridDim.x * blockDim.x) {
    int r = i / cols, c = i - r * cols;
    out[(size_t)c * rows + r] = f2bf(in[i]);
  }
}

// ---------------------------------------------------------------------------
// RMSNorm: one block per token row (1024 cols), bf16 output (GEMM A operand)
// ---------------------------------------------------------------------------
__global__ void k_rmsnorm_bf16(const float* __restrict__ x,
                               const float* __restrict__ w,
                               unsigned short* __restrict__ out) {
  const int row = blockIdx.x, tid = threadIdx.x;
  const float4 v = reinterpret_cast<const float4*>(x + (size_t)row * Ddim)[tid];
  __shared__ float red[256];
  red[tid] = v.x * v.x + v.y * v.y + v.z * v.z + v.w * v.w;
  __syncthreads();
#pragma unroll
  for (int o = 128; o > 0; o >>= 1) {
    if (tid < o) red[tid] += red[tid + o];
    __syncthreads();
  }
  const float inv = rsqrtf(red[0] * (1.0f / Ddim) + 1e-6f);
  const int c = tid * 4;
  out[(size_t)row * Ddim + c + 0] = f2bf(v.x * inv * w[c + 0]);
  out[(size_t)row * Ddim + c + 1] = f2bf(v.y * inv * w[c + 1]);
  out[(size_t)row * Ddim + c + 2] = f2bf(v.z * inv * w[c + 2]);
  out[(size_t)row * Ddim + c + 3] = f2bf(v.w * inv * w[c + 3]);
}

// ---------------------------------------------------------------------------
// gx[t][n][g][b][k] = sum_h xn[b,t, n*256+h] * W[g,n,k,h] + bias[g,n,k]
// Per wave: 32 tokens x 64 cols of one head; B frags reused by both A frags.
// ---------------------------------------------------------------------------
__global__ void __launch_bounds__(256, 1)
k_gemm_gx(const unsigned short* __restrict__ xn,
          const unsigned short* __restrict__ Wb,
          const float* __restrict__ bias, float* __restrict__ gx) {
  const int wave = threadIdx.x >> 5, lane = threadIdx.x & 31;
  const int task = blockIdx.x * 8 + wave;     // 512*4*16 = 32768 tasks
  const int n64 = task & 15, head = (task >> 4) & 3, mt = task >> 6;
  const int m0 = mt * 32;

  v8f acc[2][4];
#pragma unroll
  for (int s = 0; s < 2; ++s)
#pragma unroll
    for (int j = 0; j < 4; ++j) acc[s][j] = zero8();

#pragma unroll
  for (int kb = 0; kb < 8; ++kb) {
    const int k0 = kb * 32;
    const v16bf a0 = frag16(xn, m0, Ddim, head * DH + k0);
    const v16bf a1 = frag16(xn, m0 + 16, Ddim, head * DH + k0);
#pragma unroll
    for (int j = 0; j < 4; ++j) {
      const int ncol0 = n64 * 64 + j * 16;          // 0..1023 within head
      const int g = ncol0 >> 8, kc = ncol0 & 255;
      const v16bf b =
          frag16(Wb + (size_t)((g * NH + head) * DH) * DH, kc, DH, k0);
      acc[0][j] = __builtin_amdgcn_wmma_f32_16x16x32_bf16(
          false, a0, false, b, (short)0, acc[0][j], false, false);
      acc[1][j] = __builtin_amdgcn_wmma_f32_16x16x32_bf16(
          false, a1, false, b, (short)0, acc[1][j], false, false);
    }
  }
  const int mhi = (lane >> 4) << 3;
#pragma unroll
  for (int s = 0; s < 2; ++s)
#pragma unroll
    for (int j = 0; j < 4; ++j) {
      const int ncol = n64 * 64 + j * 16 + (lane & 15);
      const int g = ncol >> 8, k = ncol & 255;
      const float bv = bias[(g * NH + head) * DH + k];
#pragma unroll
      for (int r = 0; r < 8; ++r) {
        const int tok = m0 + s * 16 + r + mhi;
        const int b_ix = tok >> 11, t = tok & 2047;
        gx[((((size_t)t * NH + head) * 4 + g) * Bdim + b_ix) * DH + k] =
            acc[s][j][r] + bv;
      }
    }
}

// ---------------------------------------------------------------------------
// Recurrent scan: one workgroup per head (8 waves). h lives in LDS as the
// zero-padded 16x256 bf16 A-operand; R streamed from L2; c/n/m in registers.
// gx tile for step t+1 is prefetched into LDS with global_load_async_to_lds
// while step t's WMMA work runs (double-buffered, per-thread self-consumed).
// ---------------------------------------------------------------------------
__global__ void __launch_bounds__(256, 1)
k_scan(const float* __restrict__ gx, const unsigned short* __restrict__ Rb,
       float* __restrict__ hs) {
  const int head = blockIdx.x;
  const int tid = threadIdx.x, lane = tid & 31, wave = tid >> 5;
  __shared__ unsigned short hbf[16 * DH];       // A operand staging (8 KB)
  __shared__ float pre[4 * 16 * DH];            // recurrent pre-acts (64 KB)
  __shared__ float gxs[2][4 * Bdim * DH];       // gx double buffer (64 KB)

  for (int i = tid; i < 16 * DH; i += 256) hbf[i] = 0;
  float cst[8], nst[8], mst[8];
#pragma unroll
  for (int j = 0; j < 8; ++j) { cst[j] = 0.f; nst[j] = 0.f; mst[j] = 0.f; }
  const int b_ix = (tid * 8) >> 8;   // batch row owned by this thread
  const int ks0 = (tid * 8) & 255;   // first k owned by this thread

  // Async prefetch of this thread's 32 gx floats (4 gates x 8) for step t.
  auto issue_gx = [&](int t, int buf) {
    const float* g0 = gx + ((size_t)t * NH + head) * (4 * Bdim * DH);
#pragma unroll
    for (int g = 0; g < 4; ++g) {
      const float* src = g0 + (g * Bdim + b_ix) * DH + ks0;
      unsigned dst =
          (unsigned)(uintptr_t)&gxs[buf][(g * Bdim + b_ix) * DH + ks0];
      asm volatile("global_load_async_to_lds_b128 %0, %1, off"
                   :: "v"(dst), "v"(src) : "memory");
      asm volatile("global_load_async_to_lds_b128 %0, %1, off"
                   :: "v"(dst + 16u), "v"(src + 4) : "memory");
    }
  };

  issue_gx(0, 0);          // ASYNCcnt = 8
  __syncthreads();

  for (int t = 0; t < Tdim; ++t) {
    const int buf = t & 1;
    if (t + 1 < Tdim) issue_gx(t + 1, buf ^ 1);   // overlap with GEMM below

    // --- rec = h x R  (each wave: 8 fused gate/col tiles, shared A frag) ---
    v8f acc[8];
#pragma unroll
    for (int i = 0; i < 8; ++i) acc[i] = zero8();
#pragma unroll
    for (int kb = 0; kb < 8; ++kb) {
      const int k0 = kb * 32;
      const v16bf a = *reinterpret_cast<const v16bf*>(
          &hbf[(lane & 15) * DH + k0 + ((lane >> 4) << 4)]);
#pragma unroll
      for (int i = 0; i < 8; ++i) {
        const int tt = wave * 8 + i;                 // 0..63
        const int g = tt >> 4, nc0 = (tt & 15) << 4;
        const unsigned short* bp =
            Rb + ((size_t)((g * NH + head) * DH + nc0 + (lane & 15))) * DH +
            k0 + ((lane >> 4) << 4);
        const v16bf b = *reinterpret_cast<const v16bf*>(bp);
        acc[i] = __builtin_amdgcn_wmma_f32_16x16x32_bf16(
            false, a, false, b, (short)0, acc[i], false, false);
      }
    }
    const int mhi = (lane >> 4) << 3;
#pragma unroll
    for (int i = 0; i < 8; ++i) {
      const int tt = wave * 8 + i;
      const int g = tt >> 4, nn = ((tt & 15) << 4) + (lane & 15);
#pragma unroll
      for (int r = 0; r < 8; ++r) pre[(g * 16 + r + mhi) * DH + nn] = acc[i][r];
    }

    // Drain the 8 async ops for tile t (keep t+1's 8 in flight).
    if (t + 1 < Tdim)
      asm volatile("s_wait_asynccnt 0x8" ::: "memory");
    else
      asm volatile("s_wait_asynccnt 0x0" ::: "memory");
    __syncthreads();

    // --- pointwise sLSTM cell (fp32), 8 states per thread ---
    const float* gxt = gxs[buf];
#pragma unroll
    for (int j = 0; j < 8; ++j) {
      const int k = ks0 + j;
      const float ip = gxt[(0 * Bdim + b_ix) * DH + k] + pre[(0 * 16 + b_ix) * DH + k];
      const float fp = gxt[(1 * Bdim + b_ix) * DH + k] + pre[(1 * 16 + b_ix) * DH + k];
      const float zp = gxt[(2 * Bdim + b_ix) * DH + k] + pre[(2 * 16 + b_ix) * DH + k];
      const float op = gxt[(3 * Bdim + b_ix) * DH + k] + pre[(3 * 16 + b_ix) * DH + k];
      const float mn = fmaxf(fp + mst[j], ip);
      const float ig = __expf(ip - mn);
      const float fg = __expf(fp + mst[j] - mn);
      const float cn = fg * cst[j] + ig * tanhf(zp);
      const float nn2 = fg * nst[j] + ig;
      const float hv = (cn / nn2) * (1.0f / (1.0f + __expf(-op)));
      cst[j] = cn; nst[j] = nn2; mst[j] = mn;
      hbf[b_ix * DH + k] = f2bf(hv);
      hs[(((size_t)b_ix * Tdim + t) * NH + head) * DH + k] = hv;
    }
    __syncthreads();
  }
}

// ---------------------------------------------------------------------------
// Per-head LayerNorm over DH + residual add: xres = x + GN(hs)
// ---------------------------------------------------------------------------
__global__ void k_gn_residual(const float* __restrict__ hs,
                              const float* __restrict__ gnw,
                              const float* __restrict__ x,
                              float* __restrict__ xres) {
  const int n = blockIdx.x & 3, bt = blockIdx.x >> 2, tid = threadIdx.x;
  const float v = hs[((size_t)bt * NH + n) * DH + tid];
  __shared__ float s1[256], s2[256];
  s1[tid] = v; s2[tid] = v * v;
  __syncthreads();
#pragma unroll
  for (int o = 128; o > 0; o >>= 1) {
    if (tid < o) { s1[tid] += s1[tid + o]; s2[tid] += s2[tid + o]; }
    __syncthreads();
  }
  const float mu = s1[0] * (1.0f / DH);
  const float var = s2[0] * (1.0f / DH) - mu * mu;
  const float inv = rsqrtf(var + 1e-5f);
  const size_t off = (size_t)bt * Ddim + n * DH + tid;
  xres[off] = x[off] + (v - mu) * inv * gnw[n * DH + tid];
}

// ---------------------------------------------------------------------------
// FFN1: act = silu(y @ Wg) * (y @ Wu).  Wave = 32 tokens x 64 cols; every B
// fragment feeds 2 WMMAs (two A subtiles), A fragments feed 8.
// ---------------------------------------------------------------------------
__global__ void __launch_bounds__(256, 1)
k_ffn1(const unsigned short* __restrict__ yb,
       const unsigned short* __restrict__ WgT,
       const unsigned short* __restrict__ WuT,
       unsigned short* __restrict__ act) {
  const int wave = threadIdx.x >> 5, lane = threadIdx.x & 31;
  const int task = blockIdx.x * 8 + wave;     // 512 * 43 tasks
  const int nt = task % 43, mt = task / 43;
  const int m0 = mt * 32, f0 = nt * 64;

  v8f ag[2][4], au[2][4];
#pragma unroll
  for (int s = 0; s < 2; ++s)
#pragma unroll
    for (int j = 0; j < 4; ++j) { ag[s][j] = zero8(); au[s][j] = zero8(); }

  for (int kb = 0; kb < 32; ++kb) {
    const int k0 = kb * 32;
    const v16bf a0 = frag16(yb, m0, Ddim, k0);
    const v16bf a1 = frag16(yb, m0 + 16, Ddim, k0);
#pragma unroll
    for (int j = 0; j < 4; ++j) {
      const v16bf bg = frag16(WgT, f0 + j * 16, Ddim, k0);
      ag[0][j] = __builtin_amdgcn_wmma_f32_16x16x32_bf16(
          false, a0, false, bg, (short)0, ag[0][j], false, false);
      ag[1][j] = __builtin_amdgcn_wmma_f32_16x16x32_bf16(
          false, a1, false, bg, (short)0, ag[1][j], false, false);
      const v16bf bu = frag16(WuT, f0 + j * 16, Ddim, k0);
      au[0][j] = __builtin_amdgcn_wmma_f32_16x16x32_bf16(
          false, a0, false, bu, (short)0, au[0][j], false, false);
      au[1][j] = __builtin_amdgcn_wmma_f32_16x16x32_bf16(
          false, a1, false, bu, (short)0, au[1][j], false, false);
    }
  }
  const int mhi = (lane >> 4) << 3;
#pragma unroll
  for (int s = 0; s < 2; ++s)
#pragma unroll
    for (int j = 0; j < 4; ++j) {
      const int col = f0 + j * 16 + (lane & 15);
#pragma unroll
      for (int r = 0; r < 8; ++r) {
        const float g = ag[s][j][r];
        const float sg = g / (1.0f + __expf(-g));      // silu
        act[(size_t)(m0 + s * 16 + r + mhi) * Fdim + col] =
            f2bf(sg * au[s][j][r]);
      }
    }
}

// ---------------------------------------------------------------------------
// FFN2: out = xres + act @ Wd.  Wave = 32 tokens x 64 cols, K = 2752.
// ---------------------------------------------------------------------------
__global__ void __launch_bounds__(256, 1)
k_ffn2(const unsigned short* __restrict__ act,
       const unsigned short* __restrict__ WdT,
       const float* __restrict__ xres, float* __restrict__ out) {
  const int wave = threadIdx.x >> 5, lane = threadIdx.x & 31;
  const int task = blockIdx.x * 8 + wave;     // 512 * 16 tasks
  const int nt = task & 15, mt = task >> 4;
  const int m0 = mt * 32, n0 = nt * 64;

  v8f acc[2][4];
#pragma unroll
  for (int s = 0; s < 2; ++s)
#pragma unroll
    for (int j = 0; j < 4; ++j) acc[s][j] = zero8();

  for (int kb = 0; kb < 86; ++kb) {
    const int k0 = kb * 32;
    const v16bf a0 = frag16(act, m0, Fdim, k0);
    const v16bf a1 = frag16(act, m0 + 16, Fdim, k0);
#pragma unroll
    for (int j = 0; j < 4; ++j) {
      const v16bf b = frag16(WdT, n0 + j * 16, Fdim, k0);
      acc[0][j] = __builtin_amdgcn_wmma_f32_16x16x32_bf16(
          false, a0, false, b, (short)0, acc[0][j], false, false);
      acc[1][j] = __builtin_amdgcn_wmma_f32_16x16x32_bf16(
          false, a1, false, b, (short)0, acc[1][j], false, false);
    }
  }
  const int mhi = (lane >> 4) << 3;
#pragma unroll
  for (int s = 0; s < 2; ++s)
#pragma unroll
    for (int j = 0; j < 4; ++j) {
      const int col = n0 + j * 16 + (lane & 15);
#pragma unroll
      for (int r = 0; r < 8; ++r) {
        const size_t off = (size_t)(m0 + s * 16 + r + mhi) * Ddim + col;
        out[off] = xres[off] + acc[s][j][r];
      }
    }
}

// ---------------------------------------------------------------------------
extern "C" void kernel_launch(void* const* d_in, const int* in_sizes, int n_in,
                              void* d_out, int out_size, void* d_ws,
                              size_t ws_size, hipStream_t stream) {
  (void)in_sizes; (void)n_in; (void)out_size; (void)ws_size;
  const float* x       = (const float*)d_in[0];
  const float* ln1_w   = (const float*)d_in[1];
  const float* W_gates = (const float*)d_in[2];
  const float* R_gates = (const float*)d_in[3];
  const float* b_gates = (const float*)d_in[4];
  const float* gn_w    = (const float*)d_in[5];
  const float* ln2_w   = (const float*)d_in[6];
  const float* Wg      = (const float*)d_in[7];
  const float* Wu      = (const float*)d_in[8];
  const float* Wd      = (const float*)d_in[9];
  float* out = (float*)d_out;

  // workspace carve-out (256B aligned regions)
  char* w = (char*)d_ws;
  auto carve = [&](size_t bytes) {
    char* p = w;
    w += (bytes + 255) & ~(size_t)255;
    return p;
  };
  unsigned short* xn_bf = (unsigned short*)carve((size_t)NT * Ddim * 2);
  unsigned short* y_bf  = (unsigned short*)carve((size_t)NT * Ddim * 2);
  unsigned short* act_bf= (unsigned short*)carve((size_t)NT * Fdim * 2);
  unsigned short* Wgb   = (unsigned short*)carve((size_t)4 * NH * DH * DH * 2);
  unsigned short* Rb    = (unsigned short*)carve((size_t)4 * NH * DH * DH * 2);
  unsigned short* WgT   = (unsigned short*)carve((size_t)Ddim * Fdim * 2);
  unsigned short* WuT   = (unsigned short*)carve((size_t)Ddim * Fdim * 2);
  unsigned short* WdT   = (unsigned short*)carve((size_t)Ddim * Fdim * 2);
  float* gx   = (float*)carve((size_t)Tdim * NH * 4 * Bdim * DH * 4);
  float* hs   = (float*)carve((size_t)NT * Ddim * 4);
  float* xres = (float*)carve((size_t)NT * Ddim * 4);

  // weight prep (bf16, GEMM-friendly layouts)
  k_cvt_bf16<<<1024, 256, 0, stream>>>(W_gates, Wgb, 4 * NH * DH * DH);
  k_cvt_bf16<<<1024, 256, 0, stream>>>(R_gates, Rb, 4 * NH * DH * DH);
  k_cvt_bf16_T<<<4096, 256, 0, stream>>>(Wg, WgT, Ddim, Fdim);
  k_cvt_bf16_T<<<4096, 256, 0, stream>>>(Wu, WuT, Ddim, Fdim);
  k_cvt_bf16_T<<<4096, 256, 0, stream>>>(Wd, WdT, Fdim, Ddim);

  // sLSTM path
  k_rmsnorm_bf16<<<NT, 256, 0, stream>>>(x, ln1_w, xn_bf);
  k_gemm_gx<<<4096, 256, 0, stream>>>(xn_bf, Wgb, b_gates, gx);
  k_scan<<<NH, 256, 0, stream>>>(gx, Rb, hs);
  k_gn_residual<<<NT * NH, 256, 0, stream>>>(hs, gn_w, x, xres);

  // FFN path
  k_rmsnorm_bf16<<<NT, 256, 0, stream>>>(xres, ln2_w, y_bf);
  k_ffn1<<<2752, 256, 0, stream>>>(y_bf, WgT, WuT, act_bf);
  k_ffn2<<<1024, 256, 0, stream>>>(act_bf, WdT, xres, out);
}
